// IntenGNN_29016799052510
// MI455X (gfx1250) — matrix operand order repaired and checked by reading, not compiled, not run on hardware
//
#include <hip/hip_runtime.h>
#include <hip/hip_bf16.h>
#include <stdint.h>
#include <math.h>

#define N_NODES 800000
#define N_FRAGS 100000
#define N_ROOTS 2000
#define NFEAT   86
#define HID     128
#define OUTD    13
#define MLP_IN  397

typedef __attribute__((ext_vector_type(16))) __bf16 v16b;
typedef __attribute__((ext_vector_type(8)))  float  v8f;

// ---------- helpers ----------

__device__ __forceinline__ uint32_t f2bf_pack(float a, float b) {
  uint32_t ua = __float_as_uint(a);
  uint32_t ub = __float_as_uint(b);
  ua = (ua + 0x7FFFu + ((ua >> 16) & 1u)) >> 16;   // round-to-nearest-even
  ub = (ub + 0x7FFFu + ((ub >> 16) & 1u)) >> 16;
  return (ua & 0xFFFFu) | (ub << 16);
}

__device__ __forceinline__ v8f wmma_bf16x32(const uint32_t* a, const uint32_t* b, v8f c) {
  union U { uint32_t u[8]; v16b v; } A, B;
#pragma unroll
  for (int i = 0; i < 8; ++i) { A.u[i] = a[i]; B.u[i] = b[i]; }
  return __builtin_amdgcn_wmma_f32_16x16x32_bf16(false, A.v, false, B.v, (short)0, c,
                                                 false, false);
}

// A-matrix (16x32 bf16) K index for VGPR v, lane-half h:  pairs at kbase, kbase+1
__device__ __forceinline__ int a_kbase(int v, int h) { return 2*v + (v >= 4 ? 8 : 0) + 8*h; }
// A pair index inside a 32-K tile (pair-packed storage)
__device__ __forceinline__ int a_pidx(int v, int h) { return v + (v >= 4 ? 4 : 0) + 4*h; }
// B-matrix (32x16 bf16) pair index for VGPR v, lane-half h
__device__ __forceinline__ int b_pidx(int v, int h) { return v + (v >= 4 ? 4 : 0) + 4*h; }

// CDNA5 async global->LDS DMA (bypasses VGPRs, tracked by ASYNCcnt).
// LDS byte address = low 32 bits of the generic shared-space pointer.
__device__ __forceinline__ void async_g2l_b128(void* lds_dst, const void* gsrc) {
  asm volatile("global_load_async_to_lds_b128 %0, %1, off"
               :: "v"((uint32_t)(uintptr_t)lds_dst),
                  "v"((uint64_t)(uintptr_t)gsrc)
               : "memory");
}
__device__ __forceinline__ void wait_async0() {
  asm volatile("s_wait_asynccnt 0" ::: "memory");
}

// ---------- kernels ----------

__global__ void k_zero(float* __restrict__ p, int n) {
  int i = blockIdx.x * 256 + threadIdx.x;
  if (i < n) p[i] = 0.f;
}

__global__ void k_counts(const int* __restrict__ seg, float* __restrict__ counts) {
  int i = blockIdx.x * 256 + threadIdx.x;
  if (i < N_NODES) atomicAdd(&counts[seg[i]], 1.0f);
}

__global__ void k_root(const float* __restrict__ root_repr, const float* __restrict__ W_root,
                       const float* __restrict__ b_root, float* __restrict__ root_emb) {
  int i = blockIdx.x * 256 + threadIdx.x;
  if (i >= N_ROOTS * HID) return;
  int r = i / HID, c = i % HID;
  float s = b_root[c];
  const float* rp = root_repr + (size_t)r * NFEAT;
  for (int k = 0; k < NFEAT; ++k) s += rp[k] * W_root[k * HID + c];
  root_emb[i] = s > 0.f ? s : 0.f;
}

// Pack all GEMM B matrices as bf16 K-pair-packed dwords (K padded with zeros).
__global__ void k_pack(const float* __restrict__ W_in, const float* __restrict__ W_g1,
                       const float* __restrict__ W_g2, const float* __restrict__ W_m1,
                       const float* __restrict__ W_m2, const float* __restrict__ W_out,
                       const float* __restrict__ W_attn, const float* __restrict__ b_out,
                       const float* __restrict__ b_attn,
                       uint32_t* __restrict__ pWin, uint32_t* __restrict__ pWg1,
                       uint32_t* __restrict__ pWg2, uint32_t* __restrict__ pWm1,
                       uint32_t* __restrict__ pWm2, uint32_t* __restrict__ pWo,
                       float* __restrict__ pbo) {
  int i = blockIdx.x * 256 + threadIdx.x;
  const int sWin = 48 * 128, sWg = 64 * 128, sWm1 = 208 * 128, sWo = 64 * 32;
  if (i < sWin) {
    int p = i / 128, n = i % 128;
    float a = (2*p     < NFEAT) ? W_in[(2*p)     * HID + n] : 0.f;
    float b = (2*p + 1 < NFEAT) ? W_in[(2*p + 1) * HID + n] : 0.f;
    pWin[i] = f2bf_pack(a, b); return;
  }
  i -= sWin;
  if (i < sWg) { int p = i/128, n = i%128;
    pWg1[i] = f2bf_pack(W_g1[2*p*HID + n], W_g1[(2*p+1)*HID + n]); return; }
  i -= sWg;
  if (i < sWg) { int p = i/128, n = i%128;
    pWg2[i] = f2bf_pack(W_g2[2*p*HID + n], W_g2[(2*p+1)*HID + n]); return; }
  i -= sWg;
  if (i < sWm1) {
    int p = i / 128, n = i % 128;
    float a = (2*p     < MLP_IN) ? W_m1[(2*p)     * HID + n] : 0.f;
    float b = (2*p + 1 < MLP_IN) ? W_m1[(2*p + 1) * HID + n] : 0.f;
    pWm1[i] = f2bf_pack(a, b); return;
  }
  i -= sWm1;
  if (i < sWg) { int p = i/128, n = i%128;
    pWm2[i] = f2bf_pack(W_m2[2*p*HID + n], W_m2[(2*p+1)*HID + n]); return; }
  i -= sWg;
  if (i < sWo) {
    int p = i / 32, n = i % 32;
    int c = n & 15;
    float a = 0.f, b = 0.f;
    if (c < OUTD) {
      const float* W = (n >= 16) ? W_attn : W_out;
      a = W[(2*p)     * OUTD + c];
      b = W[(2*p + 1) * OUTD + c];
    }
    pWo[i] = f2bf_pack(a, b); return;
  }
  i -= sWo;
  if (i < 32) {
    int c = i & 15;
    float v = 0.f;
    if (c < OUTD) v = (i >= 16) ? b_attn[c] : b_out[c];
    pbo[i] = v;
  }
}

// Node chain: h1 = A@Win+b ; h2 = relu(h1@Wg1+b) ; h3 = relu(h2@Wg2+b); segment-sum h3.
// One 16-node tile per wave; 8 waves / block. LDS: packed weights + per-wave f32 stage.
__global__ __launch_bounds__(256)
void k_nodes(const float* __restrict__ node_h, const int* __restrict__ seg,
             const uint32_t* __restrict__ gWin,   // Win|Wg1|Wg2 contiguous (22528 dwords)
             const float* __restrict__ b_in,
             const float* __restrict__ b_g1, const float* __restrict__ b_g2,
             float* __restrict__ frag_sums) {
  extern __shared__ char smem[];
  uint32_t* sWin = (uint32_t*)smem;          // 48*128
  uint32_t* sWg1 = sWin + 48 * 128;          // 64*128
  uint32_t* sWg2 = sWg1 + 64 * 128;          // 64*128
  float*    sB   = (float*)(sWg2 + 64 * 128);// 3*128
  float*    stage = sB + 3 * 128;            // 8 * 16*132

  int tid = threadIdx.x;
  // async DMA the 3 packed weight blocks (contiguous, 5632 x 16B) straight into LDS
  for (int i = tid; i < 22528 / 4; i += 256)
    async_g2l_b128(sWin + 4 * i, gWin + 4 * i);
  if (tid < 128) { sB[tid] = b_in[tid]; sB[128 + tid] = b_g1[tid]; sB[256 + tid] = b_g2[tid]; }
  wait_async0();
  __syncthreads();

  int wave = tid >> 5, lane = tid & 31;
  int m = lane & 15, half = lane >> 4;
  int nodeBase = blockIdx.x * 128 + wave * 16;
  float* stg = stage + wave * (16 * 132);

  // --- layer-1 A fragments straight from global (bf16 convert on the fly) ---
  uint32_t a1[3][8];
  const float* rowp = node_h + (size_t)(nodeBase + m) * NFEAT;
#pragma unroll
  for (int kt = 0; kt < 3; ++kt)
#pragma unroll
    for (int v = 0; v < 8; ++v) {
      int kk = kt * 32 + a_kbase(v, half);
      if (kk < NFEAT) {            // NFEAT even => pair fully valid or fully padded
        float2 f = *(const float2*)(rowp + kk);
        a1[kt][v] = f2bf_pack(f.x, f.y);
      } else a1[kt][v] = 0u;
    }

  // --- layer 1 (no relu), stage f32 ---
#pragma unroll
  for (int nt = 0; nt < 8; ++nt) {
    float bv = sB[nt * 16 + m];
    v8f acc;
#pragma unroll
    for (int j = 0; j < 8; ++j) acc[j] = bv;
#pragma unroll
    for (int kt = 0; kt < 3; ++kt) {
      uint32_t bb[8];
#pragma unroll
      for (int v = 0; v < 8; ++v)
        bb[v] = sWin[(kt * 16 + b_pidx(v, half)) * 128 + nt * 16 + m];
      acc = wmma_bf16x32(a1[kt], bb, acc);
    }
#pragma unroll
    for (int j = 0; j < 8; ++j) stg[(j + 8 * half) * 132 + nt * 16 + m] = acc[j];
  }

  // --- repack stage -> A2 ---
  uint32_t a2[4][8];
#pragma unroll
  for (int kt = 0; kt < 4; ++kt)
#pragma unroll
    for (int v = 0; v < 8; ++v) {
      int kk = kt * 32 + a_kbase(v, half);
      float2 f = *(const float2*)(stg + m * 132 + kk);
      a2[kt][v] = f2bf_pack(f.x, f.y);
    }

  // --- layer 2 (relu), stage ---
#pragma unroll
  for (int nt = 0; nt < 8; ++nt) {
    float bv = sB[128 + nt * 16 + m];
    v8f acc;
#pragma unroll
    for (int j = 0; j < 8; ++j) acc[j] = bv;
#pragma unroll
    for (int kt = 0; kt < 4; ++kt) {
      uint32_t bb[8];
#pragma unroll
      for (int v = 0; v < 8; ++v)
        bb[v] = sWg1[(kt * 16 + b_pidx(v, half)) * 128 + nt * 16 + m];
      acc = wmma_bf16x32(a2[kt], bb, acc);
    }
#pragma unroll
    for (int j = 0; j < 8; ++j) {
      float r = acc[j] > 0.f ? acc[j] : 0.f;
      stg[(j + 8 * half) * 132 + nt * 16 + m] = r;
    }
  }

  // --- repack -> A3 ---
  uint32_t a3[4][8];
#pragma unroll
  for (int kt = 0; kt < 4; ++kt)
#pragma unroll
    for (int v = 0; v < 8; ++v) {
      int kk = kt * 32 + a_kbase(v, half);
      float2 f = *(const float2*)(stg + m * 132 + kk);
      a3[kt][v] = f2bf_pack(f.x, f.y);
    }

  // --- layer 3 (relu) fused with sorted-run segment accumulation ---
  int segs[8];
#pragma unroll
  for (int j = 0; j < 8; ++j) segs[j] = seg[nodeBase + 8 * half + j];

#pragma unroll
  for (int nt = 0; nt < 8; ++nt) {
    float bv = sB[256 + nt * 16 + m];
    v8f acc;
#pragma unroll
    for (int j = 0; j < 8; ++j) acc[j] = bv;
#pragma unroll
    for (int kt = 0; kt < 4; ++kt) {
      uint32_t bb[8];
#pragma unroll
      for (int v = 0; v < 8; ++v)
        bb[v] = sWg2[(kt * 16 + b_pidx(v, half)) * 128 + nt * 16 + m];
      acc = wmma_bf16x32(a3[kt], bb, acc);
    }
    int col = nt * 16 + m;
    float run = acc[0] > 0.f ? acc[0] : 0.f;
    int s = segs[0];
#pragma unroll
    for (int j = 1; j < 8; ++j) {
      float v = acc[j] > 0.f ? acc[j] : 0.f;
      if (segs[j] == s) run += v;
      else {
        atomicAdd(&frag_sums[(size_t)s * HID + col], run);
        s = segs[j]; run = v;
      }
    }
    atomicAdd(&frag_sums[(size_t)s * HID + col], run);
  }
}

// Fragment MLP: cat(397) -> relu -> relu -> sigmoid/attn; 16 fragments per wave.
__global__ __launch_bounds__(128)
void k_frags(const float* __restrict__ frag_sums, const float* __restrict__ counts,
             const float* __restrict__ root_emb, const int* __restrict__ ind_maps,
             const int* __restrict__ broken, const uint32_t* __restrict__ gWm1,
             const uint32_t* __restrict__ gWm2,   // Wm2|Wo contiguous (10240 dwords)
             const float* __restrict__ b_m1, const float* __restrict__ b_m2,
             const float* __restrict__ gbo, float* __restrict__ out) {
  extern __shared__ char smem[];
  uint32_t* sWm2 = (uint32_t*)smem;           // 64*128
  uint32_t* sWo  = sWm2 + 64 * 128;           // 64*32
  float* sBm1 = (float*)(sWo + 64 * 32);      // 128
  float* sBm2 = sBm1 + 128;                   // 128
  float* sBo  = sBm2 + 128;                   // 32
  uint32_t* stageAll = (uint32_t*)(sBo + 32); // 4 * 16*209

  int tid = threadIdx.x;
  // async DMA Wm2|Wo (contiguous, 2560 x 16B) into LDS
  for (int i = tid; i < 10240 / 4; i += 128)
    async_g2l_b128(sWm2 + 4 * i, gWm2 + 4 * i);
  if (tid < 128) { sBm1[tid] = b_m1[tid]; sBm2[tid] = b_m2[tid]; }
  if (tid < 32) sBo[tid] = gbo[tid];
  wait_async0();
  __syncthreads();

  int wave = tid >> 5, lane = tid & 31;
  int m = lane & 15, half = lane >> 4;
  int fragBase = (blockIdx.x * 4 + wave) * 16;
  if (fragBase >= N_FRAGS) return;
  uint32_t* stg = stageAll + wave * (16 * 209);

  // --- build concat rows pair-packed bf16: [er | er-avg | avg | onehot | 0pad] (416) ---
  for (int idx = lane; idx < 16 * 208; idx += 32) {
    int row = idx / 208, p = idx % 208;
    int f = fragBase + row;
    int k = 2 * p;
    float vals[2];
#pragma unroll
    for (int t = 0; t < 2; ++t) {
      int kk = k + t;
      float v = 0.f;
      if (kk < 128) {
        v = root_emb[(size_t)ind_maps[f] * HID + kk];
      } else if (kk < 256) {
        int c = kk - 128;
        float cnt = counts[f]; if (cnt < 1.f) cnt = 1.f;
        v = root_emb[(size_t)ind_maps[f] * HID + c] - frag_sums[(size_t)f * HID + c] / cnt;
      } else if (kk < 384) {
        int c = kk - 256;
        float cnt = counts[f]; if (cnt < 1.f) cnt = 1.f;
        v = frag_sums[(size_t)f * HID + c] / cnt;
      } else if (kk < MLP_IN) {
        int bi = broken[f]; if (bi < 0) bi = 0; if (bi > 12) bi = 12;
        v = ((kk - 384) == bi) ? 1.f : 0.f;
      }
      vals[t] = v;
    }
    stg[row * 209 + p] = f2bf_pack(vals[0], vals[1]);
  }

  // --- M1: [16,416] x [416,128], B streamed from L2-resident global ---
  v8f h[8];
#pragma unroll
  for (int nt = 0; nt < 8; ++nt) {
    float bv = sBm1[nt * 16 + m];
    v8f acc;
#pragma unroll
    for (int j = 0; j < 8; ++j) acc[j] = bv;
#pragma unroll
    for (int kt = 0; kt < 13; ++kt) {
      uint32_t aa[8], bb[8];
#pragma unroll
      for (int v = 0; v < 8; ++v) {
        aa[v] = stg[m * 209 + kt * 16 + a_pidx(v, half)];
        bb[v] = gWm1[(size_t)(kt * 16 + b_pidx(v, half)) * 128 + nt * 16 + m];
      }
      acc = wmma_bf16x32(aa, bb, acc);
    }
#pragma unroll
    for (int j = 0; j < 8; ++j) { float r = acc[j]; h[nt][j] = r > 0.f ? r : 0.f; }
  }

  // restage hidden1 as f32 (stride 132 fits inside the 209-dword rows)
  float* fst = (float*)stg;
#pragma unroll
  for (int nt = 0; nt < 8; ++nt)
#pragma unroll
    for (int j = 0; j < 8; ++j) fst[(j + 8 * half) * 132 + nt * 16 + m] = h[nt][j];

  uint32_t a2[4][8];
#pragma unroll
  for (int kt = 0; kt < 4; ++kt)
#pragma unroll
    for (int v = 0; v < 8; ++v) {
      int kk = kt * 32 + a_kbase(v, half);
      float2 f = *(const float2*)(fst + m * 132 + kk);
      a2[kt][v] = f2bf_pack(f.x, f.y);
    }

  // --- M2: [16,128] x [128,128] + relu ---
#pragma unroll
  for (int nt = 0; nt < 8; ++nt) {
    float bv = sBm2[nt * 16 + m];
    v8f acc;
#pragma unroll
    for (int j = 0; j < 8; ++j) acc[j] = bv;
#pragma unroll
    for (int kt = 0; kt < 4; ++kt) {
      uint32_t bb[8];
#pragma unroll
      for (int v = 0; v < 8; ++v)
        bb[v] = sWm2[(kt * 16 + b_pidx(v, half)) * 128 + nt * 16 + m];
      acc = wmma_bf16x32(a2[kt], bb, acc);
    }
#pragma unroll
    for (int j = 0; j < 8; ++j) { float r = acc[j]; h[nt][j] = r > 0.f ? r : 0.f; }
  }

  // restage hidden2
#pragma unroll
  for (int nt = 0; nt < 8; ++nt)
#pragma unroll
    for (int j = 0; j < 8; ++j) fst[(j + 8 * half) * 132 + nt * 16 + m] = h[nt][j];

  uint32_t a3[4][8];
#pragma unroll
  for (int kt = 0; kt < 4; ++kt)
#pragma unroll
    for (int v = 0; v < 8; ++v) {
      int kk = kt * 32 + a_kbase(v, half);
      float2 f = *(const float2*)(fst + m * 132 + kk);
      a3[kt][v] = f2bf_pack(f.x, f.y);
    }

  // --- fused output GEMM: [16,128] x [128,32] = [sigmoid out | attn] ---
#pragma unroll
  for (int nt = 0; nt < 2; ++nt) {
    float bv = sBo[nt * 16 + m];
    v8f acc;
#pragma unroll
    for (int j = 0; j < 8; ++j) acc[j] = bv;
#pragma unroll
    for (int kt = 0; kt < 4; ++kt) {
      uint32_t bb[8];
#pragma unroll
      for (int v = 0; v < 8; ++v)
        bb[v] = sWo[(kt * 16 + b_pidx(v, half)) * 32 + nt * 16 + m];
      acc = wmma_bf16x32(a3[kt], bb, acc);
    }
    if (m < OUTD) {
#pragma unroll
      for (int j = 0; j < 8; ++j) {
        int fr = fragBase + j + 8 * half;
        float v = acc[j];
        if (nt == 0) {
          v = 1.f / (1.f + expf(-v));
          out[(size_t)fr * OUTD + m] = v;
        } else {
          out[(size_t)N_FRAGS * OUTD + (size_t)fr * OUTD + m] = v;
        }
      }
    }
  }
}

// ---------- launch ----------

extern "C" void kernel_launch(void* const* d_in, const int* in_sizes, int n_in,
                              void* d_out, int out_size, void* d_ws, size_t ws_size,
                              hipStream_t stream) {
  const float* node_h    = (const float*)d_in[0];
  const int*   seg       = (const int*)  d_in[1];
  const float* root_repr = (const float*)d_in[2];
  const int*   ind_maps  = (const int*)  d_in[3];
  const int*   broken    = (const int*)  d_in[4];
  const float* W_root = (const float*)d_in[5];
  const float* b_root = (const float*)d_in[6];
  const float* W_in   = (const float*)d_in[7];
  const float* b_in   = (const float*)d_in[8];
  const float* W_g1   = (const float*)d_in[9];
  const float* b_g1   = (const float*)d_in[10];
  const float* W_g2   = (const float*)d_in[11];
  const float* b_g2   = (const float*)d_in[12];
  const float* W_m1   = (const float*)d_in[13];
  const float* b_m1   = (const float*)d_in[14];
  const float* W_m2   = (const float*)d_in[15];
  const float* b_m2   = (const float*)d_in[16];
  const float* W_out  = (const float*)d_in[17];
  const float* b_out  = (const float*)d_in[18];
  const float* W_attn = (const float*)d_in[19];
  const float* b_attn = (const float*)d_in[20];

  float* ws = (float*)d_ws;
  size_t o = 0;
  float* frag_sums = ws + o;            o += (size_t)N_FRAGS * HID;
  float* counts    = ws + o;            o += N_FRAGS;
  float* root_emb  = ws + o;            o += (size_t)N_ROOTS * HID;
  uint32_t* pWin = (uint32_t*)(ws + o); o += 48 * 128;   // Win|Wg1|Wg2 contiguous
  uint32_t* pWg1 = (uint32_t*)(ws + o); o += 64 * 128;
  uint32_t* pWg2 = (uint32_t*)(ws + o); o += 64 * 128;
  uint32_t* pWm1 = (uint32_t*)(ws + o); o += 208 * 128;
  uint32_t* pWm2 = (uint32_t*)(ws + o); o += 64 * 128;   // Wm2|Wo contiguous
  uint32_t* pWo  = (uint32_t*)(ws + o); o += 64 * 32;
  float* pbo     = ws + o;              o += 32;

  int zn = N_FRAGS * HID + N_FRAGS;     // sums + counts contiguous
  k_zero<<<(zn + 255) / 256, 256, 0, stream>>>(frag_sums, zn);

  int pn = 48*128 + 64*128 + 64*128 + 208*128 + 64*128 + 64*32 + 32;
  k_pack<<<(pn + 255) / 256, 256, 0, stream>>>(W_in, W_g1, W_g2, W_m1, W_m2, W_out, W_attn,
                                               b_out, b_attn, pWin, pWg1, pWg2, pWm1, pWm2,
                                               pWo, pbo);
  k_root<<<(N_ROOTS * HID + 255) / 256, 256, 0, stream>>>(root_repr, W_root, b_root, root_emb);
  k_counts<<<(N_NODES + 255) / 256, 256, 0, stream>>>(seg, counts);

  size_t lds_nodes = (48*128 + 64*128 + 64*128) * 4 + 3*128*4 + 8 * 16*132 * 4; // 159232 B
  k_nodes<<<N_NODES / 128, 256, lds_nodes, stream>>>(node_h, seg, pWin,
                                                     b_in, b_g1, b_g2, frag_sums);

  size_t lds_frags = 64*128*4 + 64*32*4 + (128 + 128 + 32) * 4 + 4 * 16*209 * 4; // 95616 B
  int fragTiles = N_FRAGS / 16;               // 6250
  k_frags<<<(fragTiles + 3) / 4, 128, lds_frags, stream>>>(frag_sums, counts, root_emb,
                                                           ind_maps, broken, pWm1, pWm2,
                                                           b_m1, b_m2, pbo, (float*)d_out);
}
